// MoELayer_18184891532017
// MI455X (gfx1250) — matrix operand order repaired and checked by reading, not compiled, not run on hardware
//
#include <hip/hip_runtime.h>
#include <cstdint>
#include <cstddef>

#define D_MODEL 768
#define D_FF    3072
#define NEXP    8
#define T_TOK   2048
#define LN_EPS  1e-5f

typedef __attribute__((ext_vector_type(16))) __bf16 v16bf;
typedef __attribute__((ext_vector_type(8)))  float  v8f;

struct __align__(16) U4 { unsigned int x, y, z, w; };
union BFVec { v16bf v; unsigned int u[8]; };

__device__ __forceinline__ unsigned short f2bf(float f) {
  unsigned int u = __float_as_uint(f);
  u += 0x7FFFu + ((u >> 16) & 1u);          // round-to-nearest-even
  return (unsigned short)(u >> 16);
}

__device__ __forceinline__ v16bf pack_bf(const U4& lo, const U4& hi) {
  BFVec t;
  t.u[0] = lo.x; t.u[1] = lo.y; t.u[2] = lo.z; t.u[3] = lo.w;
  t.u[4] = hi.x; t.u[5] = hi.y; t.u[6] = hi.z; t.u[7] = hi.w;
  return t.v;
}

// silu with hardware v_rcp_f32 (result is bf16-rounded anyway)
__device__ __forceinline__ float silu_fast(float v) {
  return v * __builtin_amdgcn_rcpf(1.0f + __expf(-v));
}

// ---------------------------------------------------------------------------
// src: fp32 [E][R][C]  ->  dst: bf16 [E][C][R]   (tiled transpose via LDS)
// ---------------------------------------------------------------------------
__global__ __launch_bounds__(256) void transpose_convert_kernel(
    const float* __restrict__ src, unsigned short* __restrict__ dst,
    int R, int C)
{
  __shared__ float tile[32][33];
  const size_t base = (size_t)blockIdx.z * R * C;
  const int c0 = blockIdx.x * 32, r0 = blockIdx.y * 32;
  const int tx = threadIdx.x, ty = threadIdx.y;
  for (int j = ty; j < 32; j += 8)
    tile[j][tx] = src[base + (size_t)(r0 + j) * C + (c0 + tx)];
  __syncthreads();
  for (int j = ty; j < 32; j += 8)
    dst[base + (size_t)(c0 + j) * R + (r0 + tx)] = f2bf(tile[tx][j]);
}

// ---------------------------------------------------------------------------
// LayerNorm + router softmax top-2: one wave32 per token
// ---------------------------------------------------------------------------
__global__ __launch_bounds__(256) void ln_router_kernel(
    const float* __restrict__ x, const float* __restrict__ gamma,
    const float* __restrict__ beta, const float* __restrict__ rw,
    unsigned short* __restrict__ xbf, int* __restrict__ topi,
    float* __restrict__ topw)
{
  const int tid  = threadIdx.x;
  const int wave = tid >> 5, lane = tid & 31;
  const int t = blockIdx.x * 8 + wave;
  const float* xr = x + (size_t)t * D_MODEL;

  float v[24];
  #pragma unroll
  for (int i = 0; i < 24; ++i) v[i] = xr[lane + 32 * i];

  float s = 0.f;
  #pragma unroll
  for (int i = 0; i < 24; ++i) s += v[i];
  #pragma unroll
  for (int o = 16; o > 0; o >>= 1) s += __shfl_xor(s, o, 32);
  const float mu = s * (1.0f / D_MODEL);

  float q = 0.f;
  #pragma unroll
  for (int i = 0; i < 24; ++i) { float d = v[i] - mu; q += d * d; }
  #pragma unroll
  for (int o = 16; o > 0; o >>= 1) q += __shfl_xor(q, o, 32);
  const float rstd = rsqrtf(q * (1.0f / D_MODEL) + LN_EPS);

  float lg[8];
  #pragma unroll
  for (int e = 0; e < 8; ++e) lg[e] = 0.f;
  #pragma unroll
  for (int i = 0; i < 24; ++i) {
    const int d = lane + 32 * i;
    const float yv = (v[i] - mu) * rstd * gamma[d] + beta[d];
    xbf[(size_t)t * D_MODEL + d] = f2bf(yv);
    const float4* rp = (const float4*)(rw + d * 8);
    const float4 r0 = rp[0], r1 = rp[1];
    lg[0] += yv * r0.x; lg[1] += yv * r0.y; lg[2] += yv * r0.z; lg[3] += yv * r0.w;
    lg[4] += yv * r1.x; lg[5] += yv * r1.y; lg[6] += yv * r1.z; lg[7] += yv * r1.w;
  }
  #pragma unroll
  for (int e = 0; e < 8; ++e) {
    float u = lg[e];
    #pragma unroll
    for (int o = 16; o > 0; o >>= 1) u += __shfl_xor(u, o, 32);
    lg[e] = u;
  }
  if (lane == 0) {
    int i0 = 0; float m0 = lg[0];
    #pragma unroll
    for (int e = 1; e < 8; ++e) if (lg[e] > m0) { m0 = lg[e]; i0 = e; }
    int i1 = -1; float m1 = -3.4e38f;
    #pragma unroll
    for (int e = 0; e < 8; ++e) if (e != i0 && lg[e] > m1) { m1 = lg[e]; i1 = e; }
    const float e1 = __expf(m1 - m0);           // top-2 renormalized softmax
    const float inv = 1.0f / (1.0f + e1);
    topi[2 * t] = i0; topi[2 * t + 1] = i1;
    topw[2 * t] = inv; topw[2 * t + 1] = e1 * inv;
  }
}

// ---------------------------------------------------------------------------
// Deterministic per-expert compaction via wave ballot (no atomics)
// ---------------------------------------------------------------------------
__global__ void build_lists_kernel(const int* __restrict__ topi,
                                   int* __restrict__ counts,
                                   int* __restrict__ idxl)
{
  const int e = blockIdx.x;
  const int l = threadIdx.x;
  int cnt = 0;
  for (int base = 0; base < 2 * T_TOK; base += 32) {
    const int a = base + l;
    const bool m = (topi[a] == e);
    const unsigned mask = (unsigned)__ballot(m);   // wave32: low 32 bits
    if (m) {
      const int off = __popc(mask & ((1u << l) - 1u));
      idxl[e * T_TOK + cnt + off] = a;
    }
    cnt += __popc(mask);
  }
  if (l == 0) counts[e] = cnt;
}

// ---------------------------------------------------------------------------
// Fused expert FFN: Y[16,768] = silu(X[16,768] @ W1e) @ W2e, via bf16 WMMA
// block = 256 threads (8 waves), 16 gathered tokens per block
// ---------------------------------------------------------------------------
__global__ __launch_bounds__(256) void ffn_kernel(
    const unsigned short* __restrict__ xbf,   // [T][D] bf16
    const unsigned short* __restrict__ w1t,   // [E][F][D] bf16 (transposed)
    const unsigned short* __restrict__ w2t,   // [E][D][F] bf16 (transposed)
    const int* __restrict__ counts,
    const int* __restrict__ idxl,
    float* __restrict__ y)                    // [T*2][D] fp32
{
  __shared__ __align__(16) unsigned int s_x[16 * 388]; // 16 rows, 776 bf16 (pad)
  __shared__ __align__(16) unsigned int s_h[16 * 68];  // 16 rows, 136 bf16 (pad)
  __shared__ int s_a[16];

  const int e    = blockIdx.y;
  const int tile = blockIdx.x;
  const int cnt  = counts[e];
  if (tile * 16 >= cnt) return;               // uniform per block -> EXEC stays full

  const int tid  = threadIdx.x;
  const int wave = tid >> 5;
  const int lane = tid & 31;
  const int nl   = lane & 15;
  const int kh   = lane >> 4;

  if (tid < 16) {
    const int idx = tile * 16 + tid;
    s_a[tid] = (idx < cnt) ? idxl[e * T_TOK + idx] : -1;
  }
  __syncthreads();

  const unsigned int* x32 = (const unsigned int*)xbf;
  for (int i = tid; i < 16 * 384; i += 256) {
    const int r = i / 384, c = i - r * 384;
    const int a = s_a[r];
    const int t = (a >= 0) ? (a >> 1) : 0;
    s_x[r * 388 + c] = x32[t * 384 + c];
  }
  __syncthreads();

  const U4* w1q = (const U4*)w1t + (size_t)e * ((size_t)D_MODEL * D_FF / 8);
  const U4* w2q = (const U4*)w2t + (size_t)e * ((size_t)D_MODEL * D_FF / 8);
  const U4* xq  = (const U4*)s_x;
  const U4* hq  = (const U4*)s_h;

  v8f acc[6];
  #pragma unroll
  for (int j = 0; j < 6; ++j) acc[j] = v8f{};

  #pragma unroll 1
  for (int f0 = 0; f0 < D_FF; f0 += 128) {
    // GEMM1: this wave's H tile, cols f = f0 + wave*16 + nl
    v8f ch = v8f{};
    const int frow = f0 + wave * 16 + nl;
    #pragma unroll 2
    for (int k0 = 0; k0 < D_MODEL; k0 += 32) {
      // A (16x32 bf16): lane row=nl, interleaved K groups of 8 (lane-half kh)
      const U4 alo = xq[nl * 97 + (k0 >> 3) + kh];
      const U4 ahi = xq[nl * 97 + (k0 >> 3) + kh + 2];
      // B (32x16 bf16): lane col=nl, interleaved K groups of 8 (lane-half kh)
      const U4 blo = w1q[frow * 96 + (k0 >> 3) + kh];
      const U4 bhi = w1q[frow * 96 + (k0 >> 3) + kh + 2];
      ch = __builtin_amdgcn_wmma_f32_16x16x32_bf16(
          false, pack_bf(alo, ahi), false, pack_bf(blo, bhi),
          (short)0, ch, false, false);
    }

    __syncthreads();   // previous chunk's H fully consumed
    {
      unsigned short* hs = (unsigned short*)s_h;
      #pragma unroll
      for (int j = 0; j < 8; ++j) {
        const int m = kh * 8 + j;                     // C/D row layout
        hs[m * 136 + wave * 16 + nl] = f2bf(silu_fast(ch[j]));
      }
    }
    __syncthreads();

    // GEMM2: Y[16, wave*96 + jt*16 ..] += H[16,128] @ W2[f0.., :]
    #pragma unroll 1
    for (int jt = 0; jt < 6; ++jt) {
      const int dcol = wave * 96 + jt * 16 + nl;
      #pragma unroll
      for (int kk = 0; kk < 128; kk += 32) {
        const U4 alo = hq[nl * 17 + (kk >> 3) + kh];
        const U4 ahi = hq[nl * 17 + (kk >> 3) + kh + 2];
        const int fb = f0 + kk;
        const U4 blo = w2q[dcol * 384 + (fb >> 3) + kh];
        const U4 bhi = w2q[dcol * 384 + (fb >> 3) + kh + 2];
        acc[jt] = __builtin_amdgcn_wmma_f32_16x16x32_bf16(
            false, pack_bf(alo, ahi), false, pack_bf(blo, bhi),
            (short)0, acc[jt], false, false);
      }
    }
  }

  #pragma unroll
  for (int jt = 0; jt < 6; ++jt) {
    const int dcol = wave * 96 + jt * 16 + nl;
    #pragma unroll
    for (int j = 0; j < 8; ++j) {
      const int m = kh * 8 + j;
      const int a = s_a[m];
      if (a >= 0) y[(size_t)a * D_MODEL + dcol] = acc[jt][j];
    }
  }
}

// ---------------------------------------------------------------------------
// out = residual + g0*y[slot0] + g1*y[slot1]
// ---------------------------------------------------------------------------
__global__ __launch_bounds__(256) void combine_kernel(
    const float* __restrict__ resid, const float* __restrict__ topw,
    const float* __restrict__ y, float* __restrict__ out)
{
  const int t = blockIdx.y;
  const int d = blockIdx.x * 256 + threadIdx.x;
  const float g0 = topw[2 * t], g1 = topw[2 * t + 1];
  const size_t i = (size_t)t * D_MODEL + d;
  out[i] = resid[i] + g0 * y[(size_t)(2 * t) * D_MODEL + d]
                    + g1 * y[(size_t)(2 * t + 1) * D_MODEL + d];
}

// ---------------------------------------------------------------------------
extern "C" void kernel_launch(void* const* d_in, const int* in_sizes, int n_in,
                              void* d_out, int out_size, void* d_ws, size_t ws_size,
                              hipStream_t stream)
{
  const float* x     = (const float*)d_in[0];   // [2,1024,768]
  const float* gamma = (const float*)d_in[1];   // [768]
  const float* beta  = (const float*)d_in[2];   // [768]
  const float* rw    = (const float*)d_in[3];   // [768,8]
  const float* w1    = (const float*)d_in[4];   // [8,768,3072]
  const float* w2    = (const float*)d_in[5];   // [8,3072,768]
  float* out = (float*)d_out;

  char* ws = (char*)d_ws;
  const size_t WB = (size_t)NEXP * D_MODEL * D_FF * 2;      // 37,748,736 B each
  unsigned short* w1t = (unsigned short*)(ws);
  unsigned short* w2t = (unsigned short*)(ws + WB);
  unsigned short* xbf = (unsigned short*)(ws + 2 * WB);
  size_t o = 2 * WB + (size_t)T_TOK * D_MODEL * 2;          // +3,145,728
  int*   topi = (int*)(ws + o);        o += 2 * T_TOK * 4;  // 16 KB
  float* topw = (float*)(ws + o);      o += 2 * T_TOK * 4;  // 16 KB
  int*   cnts = (int*)(ws + o);        o += 256;
  int*   idxl = (int*)(ws + o);        o += (size_t)NEXP * T_TOK * 4;
  float* ybuf = (float*)(ws + o);      // [4096][768] fp32 = 12.6 MB

  transpose_convert_kernel<<<dim3(D_FF / 32, D_MODEL / 32, NEXP), dim3(32, 8),
                             0, stream>>>(w1, w1t, D_MODEL, D_FF);
  transpose_convert_kernel<<<dim3(D_MODEL / 32, D_FF / 32, NEXP), dim3(32, 8),
                             0, stream>>>(w2, w2t, D_FF, D_MODEL);
  ln_router_kernel<<<T_TOK / 8, 256, 0, stream>>>(x, gamma, beta, rw,
                                                  xbf, topi, topw);
  build_lists_kernel<<<NEXP, 32, 0, stream>>>(topi, cnts, idxl);
  ffn_kernel<<<dim3(T_TOK / 16, NEXP), 256, 0, stream>>>(xbf, w1t, w2t,
                                                         cnts, idxl, ybuf);
  combine_kernel<<<dim3(D_MODEL / 256, T_TOK), 256, 0, stream>>>(x, topw,
                                                                 ybuf, out);
}
